// HG_GNN_34059090657513
// MI455X (gfx1250) — compile-verified, matrix operand order checked by reading, not run on hardware
//
#include <hip/hip_runtime.h>
#include <hip/hip_bf16.h>

typedef __attribute__((ext_vector_type(16))) _Float16 v16h;
typedef __attribute__((ext_vector_type(8)))  _Float16 v8h;
typedef __attribute__((ext_vector_type(8)))  float    v8f;

#define N_NODES 50000
#define N_EDGES 600000
#define BS      512
#define SEQL    10
#define D       128
#define NSLOT   (BS * 11)        // 5632 = 352*16
#define NITEM   (BS * SEQL)      // 5120
#define SCORE_N 49999
#define TAB_ROWS 50000           // padded to multiple of 16

static __device__ __forceinline__ v16h load_frag(const _Float16* p) {
    // p points at (row, k0) with k0 = kstep*32 + 8*(lane>>4); two contiguous 16B chunks
    union { v16h v; v8h h[2]; } u;
    u.h[0] = *reinterpret_cast<const v8h*>(p);
    u.h[1] = *reinterpret_cast<const v8h*>(p + 16);
    return u.v;
}

static __device__ __forceinline__ v8f wmma_step(v16h a, v16h b, v8f c) {
    return __builtin_amdgcn_wmma_f32_16x16x32_f16(false, a, false, b, (short)0, c, false, false);
}

static __device__ __forceinline__ float sigmoidf_(float x) { return 1.0f / (1.0f + __expf(-x)); }

// ---------------- utility kernels ----------------
__global__ void k_zero(float* p, size_t n) {
    size_t i = (size_t)blockIdx.x * blockDim.x + threadIdx.x;
    size_t stride = (size_t)gridDim.x * blockDim.x;
    for (; i < n; i += stride) p[i] = 0.0f;
}

// transpose 128x128 f32 weight -> f16 Bt layout: dst[n*128+k] = src[(rowoff+k)*128+n]
__global__ void k_wt(const float* __restrict__ src, _Float16* __restrict__ dst, int rowoff) {
    int t = blockIdx.x * blockDim.x + threadIdx.x;
    if (t >= 128 * 128) return;
    int n = t >> 7, k = t & 127;
    dst[n * 128 + k] = (_Float16)src[(rowoff + k) * 128 + n];
}

__global__ void k_cvt_rowmajor(const float* __restrict__ src, _Float16* __restrict__ dst, int n) {
    int i = blockIdx.x * blockDim.x + threadIdx.x;
    if (i < n) dst[i] = (_Float16)src[i];
}

// item table: row j = v2e[1+j] for j < SCORE_N, last row zero pad
__global__ void k_items(const float* __restrict__ v2e, _Float16* __restrict__ tab) {
    size_t i = (size_t)blockIdx.x * blockDim.x + threadIdx.x;
    size_t stride = (size_t)gridDim.x * blockDim.x;
    size_t total = (size_t)TAB_ROWS * D;
    for (; i < total; i += stride) {
        int j = (int)(i >> 7), k = (int)(i & 127);
        tab[i] = (j < SCORE_N) ? (_Float16)v2e[(size_t)(j + 1) * D + k] : (_Float16)0.0f;
    }
}

// flag needed nodes + record slot->node  (slot = b*11 + l; l==10 is the user node)
__global__ void k_flags(const int* __restrict__ user, const int* __restrict__ seq,
                        const int* __restrict__ item_num_p, int* __restrict__ flag,
                        int* __restrict__ snode) {
    int idx = blockIdx.x * blockDim.x + threadIdx.x;
    if (idx >= NSLOT) return;
    int b = idx / 11, l = idx - b * 11;
    int node = (l < 10) ? seq[b * SEQL + l] : (user[b] + item_num_p[0]);
    flag[node] = 1;
    snode[idx] = node;
}

// filtered edge scatter: one wave per edge, 4 floats/lane, f32 atomics into agg
__global__ void k_edges(const int* __restrict__ esrc, const int* __restrict__ edst,
                        const float* __restrict__ v2e, const int* __restrict__ flag,
                        float* __restrict__ agg, float* __restrict__ deg, int E) {
    int wave = blockIdx.x * (blockDim.x >> 5) + (threadIdx.x >> 5);
    int lane = threadIdx.x & 31;
    int nwaves = gridDim.x * (blockDim.x >> 5);
    for (int e = wave; e < E; e += nwaves) {
        int d = edst[e];                       // wave-uniform
        if (!flag[d]) continue;
        int s = esrc[e];
        const float4 va = *reinterpret_cast<const float4*>(v2e + (size_t)s * D + lane * 4);
        float* ap = agg + (size_t)d * D + lane * 4;
        atomicAdd(ap + 0, va.x);
        atomicAdd(ap + 1, va.y);
        atomicAdd(ap + 2, va.z);
        atomicAdd(ap + 3, va.w);
        if (lane == 0) atomicAdd(deg + d, 1.0f);
    }
}

// materialize f16 A-operands for the SAGE GEMM (keeps the WMMA kernel spill-free)
__global__ void k_rows(const float* __restrict__ v2e, const float* __restrict__ agg,
                       const float* __restrict__ deg, const int* __restrict__ snode,
                       _Float16* __restrict__ xh, _Float16* __restrict__ nh) {
    int idx = blockIdx.x * blockDim.x + threadIdx.x;
    if (idx >= NSLOT * D) return;
    int slot = idx >> 7, k = idx & 127;
    int node = snode[slot];
    float inv = 1.0f / fmaxf(deg[node], 1.0f);
    xh[idx] = (_Float16)v2e[(size_t)node * D + k];
    nh[idx] = (_Float16)(agg[(size_t)node * D + k] * inv);
}

// ---- SAGE rows as pure dual-A f16 WMMA GEMM: e = 0.5*(relu(x@Ws + n@Wn + b) + x) ----
// grid.x = NSLOT/16 (352), block 256 = 8 waves, wave -> ntile
__global__ void k_sage(const _Float16* __restrict__ xh, const _Float16* __restrict__ nh,
                       const _Float16* __restrict__ WselfT, const _Float16* __restrict__ WneighT,
                       const float* __restrict__ bsage, const float* __restrict__ v2e,
                       const int* __restrict__ snode,
                       float* __restrict__ items_f, _Float16* __restrict__ items_h,
                       float* __restrict__ users_f, _Float16* __restrict__ users_h) {
    int mt = blockIdx.x;
    int nt = threadIdx.x >> 5, lane = threadIdx.x & 31;
    int lo = lane & 15, hi = lane >> 4;
    const _Float16* a1 = xh + (mt * 16 + lo) * D + hi * 8;
    const _Float16* a2 = nh + (mt * 16 + lo) * D + hi * 8;
    const _Float16* b1 = WselfT  + (nt * 16 + lo) * D + hi * 8;
    const _Float16* b2 = WneighT + (nt * 16 + lo) * D + hi * 8;
    v8f acc = {};
#pragma unroll
    for (int ks = 0; ks < 4; ++ks) {
        acc = wmma_step(load_frag(a1 + ks * 32), load_frag(b1 + ks * 32), acc);
        acc = wmma_step(load_frag(a2 + ks * 32), load_frag(b2 + ks * 32), acc);
    }
    int n = nt * 16 + lo;
    float bias = bsage[n];
#pragma unroll
    for (int r = 0; r < 8; ++r) {
        int slot = mt * 16 + hi * 8 + r;
        int node = snode[slot];
        float h1 = acc[r] + bias;
        h1 = h1 > 0.0f ? h1 : 0.0f;
        float e = 0.5f * (h1 + v2e[(size_t)node * D + n]);
        int b = slot / 11, l = slot - b * 11;
        if (l < 10) {
            int ri = b * SEQL + l;
            items_f[ri * D + n] = e;
            items_h[ri * D + n] = (_Float16)e;
        } else {
            users_f[b * D + n] = e;
            users_h[b * D + n] = (_Float16)e;
        }
    }
}

// avg over L with mask -> f16
__global__ void k_avg(const float* __restrict__ items_f, const int* __restrict__ mask,
                      _Float16* __restrict__ avg_h) {
    int b = blockIdx.x, k = threadIdx.x;
    float s = 0.0f, ms = 0.0f;
#pragma unroll
    for (int l = 0; l < SEQL; ++l) {
        float mm = (float)mask[b * SEQL + l];
        s  += items_f[(b * SEQL + l) * D + k] * mm;
        ms += mm;
    }
    avg_h[b * D + k] = (_Float16)(s / ms);
}

// C_f32[Mx128] = A_h[Mx128] @ Bt_h[128x128]   (grid.x = M/16, block 256 = 8 waves, wave -> ntile)
__global__ void k_gemm_plain(const _Float16* __restrict__ A, const _Float16* __restrict__ Bt,
                             float* __restrict__ C) {
    int mt = blockIdx.x;
    int nt = threadIdx.x >> 5, lane = threadIdx.x & 31;
    int lo = lane & 15, hi = lane >> 4;
    const _Float16* arow = A  + (mt * 16 + lo) * D + hi * 8;
    const _Float16* brow = Bt + (nt * 16 + lo) * D + hi * 8;
    v8f acc = {};
#pragma unroll
    for (int ks = 0; ks < 4; ++ks)
        acc = wmma_step(load_frag(arow + ks * 32), load_frag(brow + ks * 32), acc);
    int n = nt * 16 + lo;
#pragma unroll
    for (int r = 0; r < 8; ++r)
        C[(mt * 16 + hi * 8 + r) * D + n] = acc[r];
}

// out_h = f16(tanh(A@Bt))
__global__ void k_gemm_tanh(const _Float16* __restrict__ A, const _Float16* __restrict__ Bt,
                            _Float16* __restrict__ outh) {
    int mt = blockIdx.x;
    int nt = threadIdx.x >> 5, lane = threadIdx.x & 31;
    int lo = lane & 15, hi = lane >> 4;
    const _Float16* arow = A  + (mt * 16 + lo) * D + hi * 8;
    const _Float16* brow = Bt + (nt * 16 + lo) * D + hi * 8;
    v8f acc = {};
#pragma unroll
    for (int ks = 0; ks < 4; ++ks)
        acc = wmma_step(load_frag(arow + ks * 32), load_frag(brow + ks * 32), acc);
    int n = nt * 16 + lo;
#pragma unroll
    for (int r = 0; r < 8; ++r)
        outh[(mt * 16 + hi * 8 + r) * D + n] = (_Float16)tanhf(acc[r]);
}

// stage1: nh1 = tanh(pos_e @ w1a + item_e @ w1b), pos rows gathered by pos_idx
__global__ void k_stage1(const _Float16* __restrict__ items_h, const _Float16* __restrict__ pos_h,
                         const int* __restrict__ pos_idx,
                         const _Float16* __restrict__ w1at, const _Float16* __restrict__ w1bt,
                         _Float16* __restrict__ nh1h) {
    int mt = blockIdx.x;                  // 0..319
    int nt = threadIdx.x >> 5, lane = threadIdx.x & 31;
    int lo = lane & 15, hi = lane >> 4;
    int ridx = mt * 16 + lo;
    int pidx = pos_idx[ridx];
    const _Float16* a1 = pos_h   + pidx * D + hi * 8;
    const _Float16* a2 = items_h + ridx * D + hi * 8;
    const _Float16* b1 = w1at + (nt * 16 + lo) * D + hi * 8;
    const _Float16* b2 = w1bt + (nt * 16 + lo) * D + hi * 8;
    v8f acc = {};
#pragma unroll
    for (int ks = 0; ks < 4; ++ks) {
        acc = wmma_step(load_frag(a1 + ks * 32), load_frag(b1 + ks * 32), acc);
        acc = wmma_step(load_frag(a2 + ks * 32), load_frag(b2 + ks * 32), acc);
    }
    int n = nt * 16 + lo;
#pragma unroll
    for (int r = 0; r < 8; ++r)
        nh1h[(mt * 16 + hi * 8 + r) * D + n] = (_Float16)tanhf(acc[r]);
}

// out_f32 = sigmoid(A@Bt + bias[n] + Add[row/10][n])
__global__ void k_sig(const _Float16* __restrict__ A, const _Float16* __restrict__ Bt,
                      const float* __restrict__ bias, const float* __restrict__ Add,
                      float* __restrict__ outf) {
    int mt = blockIdx.x;
    int nt = threadIdx.x >> 5, lane = threadIdx.x & 31;
    int lo = lane & 15, hi = lane >> 4;
    const _Float16* arow = A  + (mt * 16 + lo) * D + hi * 8;
    const _Float16* brow = Bt + (nt * 16 + lo) * D + hi * 8;
    v8f acc = {};
#pragma unroll
    for (int ks = 0; ks < 4; ++ks)
        acc = wmma_step(load_frag(arow + ks * 32), load_frag(brow + ks * 32), acc);
    int n = nt * 16 + lo;
    float bi = bias[n];
#pragma unroll
    for (int r = 0; r < 8; ++r) {
        int row = mt * 16 + hi * 8 + r;
        float x = acc[r] + bi + Add[(row / SEQL) * D + n];
        outf[row * D + n] = sigmoidf_(x);
    }
}

// per-session betas, sess vectors, alpha mix -> seq_emb f16   (one block of 128 per b)
__global__ void k_sess(const float* __restrict__ items_f, const float* __restrict__ users_f,
                       const int* __restrict__ mask,
                       const float* __restrict__ sig1, const float* __restrict__ sig2,
                       const float* __restrict__ w2, const float* __restrict__ w4,
                       const float* __restrict__ wcw, const float* __restrict__ wcb,
                       _Float16* __restrict__ semb_h) {
    int b = blockIdx.x, k = threadIdx.x;
    __shared__ float red[128];
    __shared__ float beta1[SEQL], beta2[SEQL];
    __shared__ float alpha_sh;

    for (int l = 0; l < SEQL; ++l) {
        int r = b * SEQL + l;
        float mf = (float)mask[r];
        red[k] = sig1[r * D + k] * w2[k];
        __syncthreads();
        for (int s = 64; s > 0; s >>= 1) { if (k < s) red[k] += red[k + s]; __syncthreads(); }
        if (k == 0) beta1[l] = red[0] * mf;
        __syncthreads();
        red[k] = sig2[r * D + k] * w4[k];
        __syncthreads();
        for (int s = 64; s > 0; s >>= 1) { if (k < s) red[k] += red[k + s]; __syncthreads(); }
        if (k == 0) beta2[l] = red[0] * mf;
        __syncthreads();
    }
    float sv = 0.0f, su = 0.0f;
#pragma unroll
    for (int l = 0; l < SEQL; ++l) {
        float ie = items_f[(b * SEQL + l) * D + k];
        sv += beta1[l] * ie;
        su += beta2[l] * ie;
    }
    red[k] = sv * wcw[k] + su * wcw[D + k];
    __syncthreads();
    for (int s = 64; s > 0; s >>= 1) { if (k < s) red[k] += red[k + s]; __syncthreads(); }
    if (k == 0) alpha_sh = sigmoidf_(red[0] + wcb[0]);
    __syncthreads();
    float alpha = alpha_sh;
    float seq = users_f[b * D + k] + alpha * sv + (1.0f - alpha) * su;
    semb_h[b * D + k] = (_Float16)seq;
}

// scores = seq_emb @ v2e[1:]^T  via f16 WMMA, store-bound (102 MB f32 out)
__global__ void k_scores(const _Float16* __restrict__ semb_h, const _Float16* __restrict__ tab,
                         float* __restrict__ out) {
    int w = threadIdx.x >> 5, lane = threadIdx.x & 31;
    int lo = lane & 15, hi = lane >> 4;
    int nt = blockIdx.x * 8 + w;               // wave-uniform guard keeps EXEC all-ones
    int mt = blockIdx.y;
    if (nt >= (TAB_ROWS / 16)) return;
    const _Float16* arow = semb_h + (mt * 16 + lo) * D + hi * 8;
    const _Float16* brow = tab + (size_t)(nt * 16 + lo) * D + hi * 8;
    v8f acc = {};
#pragma unroll
    for (int ks = 0; ks < 4; ++ks)
        acc = wmma_step(load_frag(arow + ks * 32), load_frag(brow + ks * 32), acc);
    int n = nt * 16 + lo;
    if (n < SCORE_N) {
        int base = mt * 16 + hi * 8;
#pragma unroll
        for (int r = 0; r < 8; ++r)
            out[(size_t)(base + r) * SCORE_N + n] = acc[r];
    }
}

extern "C" void kernel_launch(void* const* d_in, const int* in_sizes, int n_in,
                              void* d_out, int out_size, void* d_ws, size_t ws_size,
                              hipStream_t stream) {
    const int*   user      = (const int*)d_in[0];
    const int*   seq       = (const int*)d_in[1];
    const int*   mask      = (const int*)d_in[2];
    const int*   pos_idx   = (const int*)d_in[4];
    const int*   esrc      = (const int*)d_in[5];
    const int*   edst      = (const int*)d_in[6];
    const int*   item_nump = (const int*)d_in[7];
    const float* v2e       = (const float*)d_in[8];
    const float* pos_w     = (const float*)d_in[9];
    const float* W_self    = (const float*)d_in[10];
    const float* W_neigh   = (const float*)d_in[11];
    const float* b_sage    = (const float*)d_in[12];
    const float* w1        = (const float*)d_in[13];
    const float* w2        = (const float*)d_in[14];
    const float* glu1_w    = (const float*)d_in[15];
    const float* glu1_b    = (const float*)d_in[16];
    const float* glu2_w    = (const float*)d_in[17];
    const float* w3        = (const float*)d_in[18];
    const float* w4        = (const float*)d_in[19];
    const float* glu3_w    = (const float*)d_in[20];
    const float* glu3_b    = (const float*)d_in[21];
    const float* glu4_w    = (const float*)d_in[22];
    const float* wc_w      = (const float*)d_in[23];
    const float* wc_b      = (const float*)d_in[24];
    float* out = (float*)d_out;
    char*  ws  = (char*)d_ws;

    size_t off = 0;
    auto al = [&](size_t bytes) { size_t o = off; off = (off + bytes + 255) & ~(size_t)255; return o; };

    float*    agg     = (float*)(ws + al((size_t)N_NODES * D * 4));
    float*    deg     = (float*)(ws + al((size_t)N_NODES * 4));
    int*      flag    = (int*)  (ws + al((size_t)N_NODES * 4));
    size_t zero_words = off / 4;                       // agg + deg + flag (incl. pad)
    int*      snode   = (int*)  (ws + al((size_t)NSLOT * 4));
    _Float16* wselfT  = (_Float16*)(ws + al(128 * 128 * 2));
    _Float16* wneighT = (_Float16*)(ws + al(128 * 128 * 2));
    _Float16* w1at    = (_Float16*)(ws + al(128 * 128 * 2));
    _Float16* w1bt    = (_Float16*)(ws + al(128 * 128 * 2));
    _Float16* glu1t   = (_Float16*)(ws + al(128 * 128 * 2));
    _Float16* glu2t   = (_Float16*)(ws + al(128 * 128 * 2));
    _Float16* w3t     = (_Float16*)(ws + al(128 * 128 * 2));
    _Float16* glu3t   = (_Float16*)(ws + al(128 * 128 * 2));
    _Float16* glu4t   = (_Float16*)(ws + al(128 * 128 * 2));
    _Float16* pos_h   = (_Float16*)(ws + al(200 * 128 * 2));
    _Float16* tab     = (_Float16*)(ws + al((size_t)TAB_ROWS * D * 2));
    _Float16* xh      = (_Float16*)(ws + al((size_t)NSLOT * D * 2));
    _Float16* nhrow   = (_Float16*)(ws + al((size_t)NSLOT * D * 2));
    float*    items_f = (float*)   (ws + al((size_t)NITEM * D * 4));
    _Float16* items_h = (_Float16*)(ws + al((size_t)NITEM * D * 2));
    float*    users_f = (float*)   (ws + al((size_t)BS * D * 4));
    _Float16* users_h = (_Float16*)(ws + al((size_t)BS * D * 2));
    _Float16* avg_h   = (_Float16*)(ws + al((size_t)BS * D * 2));
    float*    AG      = (float*)   (ws + al((size_t)BS * D * 4));
    float*    U4      = (float*)   (ws + al((size_t)BS * D * 4));
    _Float16* nh1_h   = (_Float16*)(ws + al((size_t)NITEM * D * 2));
    _Float16* nh3_h   = (_Float16*)(ws + al((size_t)NITEM * D * 2));
    float*    sig1    = (float*)   (ws + al((size_t)NITEM * D * 4));
    float*    sig2    = (float*)   (ws + al((size_t)NITEM * D * 4));
    _Float16* semb_h  = (_Float16*)(ws + al((size_t)BS * D * 2));

    // 1) zero graph accumulators
    k_zero<<<4096, 256, 0, stream>>>((float*)ws, zero_words);
    // 2) mark needed nodes, record slot list
    k_flags<<<(NSLOT + 255) / 256, 256, 0, stream>>>(user, seq, item_nump, flag, snode);
    // 3) filtered edge scatter (one wave per edge)
    k_edges<<<2048, 256, 0, stream>>>(esrc, edst, v2e, flag, agg, deg, N_EDGES);
    // 4) weight conversions (f16, transposed Bt layout)
    k_wt<<<64, 256, 0, stream>>>(W_self,  wselfT,  0);
    k_wt<<<64, 256, 0, stream>>>(W_neigh, wneighT, 0);
    k_wt<<<64, 256, 0, stream>>>(w1,      w1at,    0);
    k_wt<<<64, 256, 0, stream>>>(w1,      w1bt,    128);
    k_wt<<<64, 256, 0, stream>>>(glu1_w,  glu1t,   0);
    k_wt<<<64, 256, 0, stream>>>(glu2_w,  glu2t,   0);
    k_wt<<<64, 256, 0, stream>>>(w3,      w3t,     0);
    k_wt<<<64, 256, 0, stream>>>(glu3_w,  glu3t,   0);
    k_wt<<<64, 256, 0, stream>>>(glu4_w,  glu4t,   0);
    k_cvt_rowmajor<<<100, 256, 0, stream>>>(pos_w, pos_h, 200 * 128);
    k_items<<<4096, 256, 0, stream>>>(v2e, tab);
    // 5) materialize f16 A-operands for SAGE (x rows + normalized neighbor rows)
    k_rows<<<(NSLOT * D + 255) / 256, 256, 0, stream>>>(v2e, agg, deg, snode, xh, nhrow);
    // 6) SAGE rows as dual-A WMMA GEMM (fused relu + 0.5*(h1+x) epilogue)
    k_sage<<<NSLOT / 16, 256, 0, stream>>>(xh, nhrow, wselfT, wneighT, b_sage, v2e, snode,
                                           items_f, items_h, users_f, users_h);
    // 7) masked average
    k_avg<<<BS, 128, 0, stream>>>(items_f, mask, avg_h);
    // 8) GLU broadcast terms
    k_gemm_plain<<<BS / 16, 256, 0, stream>>>(avg_h,   glu2t, AG);
    k_gemm_plain<<<BS / 16, 256, 0, stream>>>(users_h, glu4t, U4);
    // 9) attention chain 1
    k_stage1<<<NITEM / 16, 256, 0, stream>>>(items_h, pos_h, pos_idx, w1at, w1bt, nh1_h);
    k_sig<<<NITEM / 16, 256, 0, stream>>>(nh1_h, glu1t, glu1_b, AG, sig1);
    // 10) attention chain 2
    k_gemm_tanh<<<NITEM / 16, 256, 0, stream>>>(items_h, w3t, nh3_h);
    k_sig<<<NITEM / 16, 256, 0, stream>>>(nh3_h, glu3t, glu3_b, U4, sig2);
    // 11) betas, session vectors, alpha mix -> seq_emb (f16)
    k_sess<<<BS, 128, 0, stream>>>(items_f, users_f, mask, sig1, sig2, w2, w4, wc_w, wc_b, semb_h);
    // 12) final store-bound scores GEMM (512 x 49999)
    dim3 g((TAB_ROWS / 16 + 7) / 8, BS / 16);
    k_scores<<<g, 256, 0, stream>>>(semb_h, tab, out);
}
